// BauAttn_90640989815033
// MI455X (gfx1250) — compile-verified, matrix operand order checked by reading, not run on hardware
//
#include <hip/hip_runtime.h>

// ---------------------------------------------------------------------------
// Bahdanau attention, MI455X (gfx1250, wave32, WMMA bf16, TDM async loads)
// L=2048 B=32 ENC=ATTN=HID=1024, M = L*B = 65536 flattened rows.
// ---------------------------------------------------------------------------

typedef unsigned int  uint_t;
typedef unsigned short ushort_t;

typedef float    __attribute__((ext_vector_type(4)))  float4v;
typedef float    __attribute__((ext_vector_type(8)))  v8f;
typedef __bf16   __attribute__((ext_vector_type(16))) v16bf;
typedef uint_t   __attribute__((ext_vector_type(4)))  uint4v;
typedef int      __attribute__((ext_vector_type(4)))  int4v;
typedef int      __attribute__((ext_vector_type(8)))  int8v;
typedef ushort_t __attribute__((ext_vector_type(4)))  ushort4v;
typedef ushort_t __attribute__((ext_vector_type(8)))  ushort8v;

#define LL   2048
#define BB   32
#define EE   1024
#define AA   1024
#define MM   (LL*BB)          // 65536
#define MT   64               // M-rows per workgroup
#define ROWB 2064u            // padded LDS row stride (1024 bf16 + 2x8B pads)
#define A_BYTES  (64u*ROWB)   // 132096
#define BCH_BYTES (32u*ROWB)  // 66048 (32-column W chunk)
#define SMEM_BYTES (A_BYTES + 2u*BCH_BYTES)  // 264192 (<320KB WGP LDS)

// Hardware V_TANH_F32 (CDNA5 TRANS op) — confirmed present last round.
#if __has_builtin(__builtin_amdgcn_tanhf)
__device__ __forceinline__ float fast_tanh(float x) { return __builtin_amdgcn_tanhf(x); }
#else
__device__ __forceinline__ float fast_tanh(float x) { return tanhf(x); }
#endif

__device__ __forceinline__ ushort_t f2bf(float f) {
    uint_t u = __float_as_uint(f);
    u = (u + 0x7FFFu + ((u >> 16) & 1u)) >> 16;   // round-to-nearest-even
    return (ushort_t)u;
}
__device__ __forceinline__ float bf2f(ushort_t h) {
    return __uint_as_float(((uint_t)h) << 16);
}

// Padded LDS offset for byte x (0..2047) inside a 2048-B logical row:
// stream layout = 1024B data, 8B pad, 1024B data, 8B pad
// (TDM pad_interval=256DW, pad_amount=2DW) -> bank stride 4 mod 64: conflict-free
__device__ __forceinline__ uint_t pad_off(uint_t x) { return x + ((x >= 1024u) ? 8u : 0u); }

// ---------------------------------------------------------------------------
// TDM: 2-D tensor load to LDS (bf16 elements, padded rows as above).
// ---------------------------------------------------------------------------
__device__ __forceinline__ void tdm_load_2d(uint_t lds_addr, const void* gptr,
                                            uint_t tensor_d0, uint_t tensor_d1,
                                            uint_t tile_d0, uint_t tile_d1,
                                            uint_t stride0_elems) {
    unsigned long long ga = (unsigned long long)gptr;
    uint4v g0;
    g0[0] = 1u;                                   // count=1, user descriptor
    g0[1] = lds_addr;                             // LDS byte address
    g0[2] = (uint_t)(ga & 0xFFFFFFFFull);         // global_addr[31:0]
    g0[3] = (uint_t)((ga >> 32) & 0x1FFFFFFull) | (2u << 30);  // addr hi | type=2
    int8v g1;
    g1[0] = (int)((1u << 16)      // data_size = 2 bytes (bf16)
                | (1u << 20)      // pad_enable
                | (7u << 22)      // pad_interval: 256 DWORDs (1024 B)
                | (1u << 25));    // pad_amount: 2 DWORDs (8 B)
    g1[1] = (int)((tensor_d0 & 0xFFFFu) << 16);
    g1[2] = (int)(((tensor_d0 >> 16) & 0xFFFFu) | ((tensor_d1 & 0xFFFFu) << 16));
    g1[3] = (int)(((tensor_d1 >> 16) & 0xFFFFu) | ((tile_d0 & 0xFFFFu) << 16));
    g1[4] = (int)(tile_d1 & 0xFFFFu);             // tile_dim2 = 0
    g1[5] = (int)stride0_elems;                   // tensor_dim0_stride[31:0]
    g1[6] = 0;
    g1[7] = 0;
    int4v gz = {0, 0, 0, 0};
#if defined(__clang_major__) && (__clang_major__ >= 23)
    int8v gz8 = {0,0,0,0,0,0,0,0};
    __builtin_amdgcn_tensor_load_to_lds(g0, g1, gz, gz, gz8, 0);
#else
    __builtin_amdgcn_tensor_load_to_lds(g0, g1, gz, gz, 0);
#endif
}

// ---------------------------------------------------------------------------
// 1) enc fp32 -> bf16 (128 MB copy that then lives in the 192 MB L2)
// ---------------------------------------------------------------------------
__global__ void convert_bf16_kernel(const float* __restrict__ in, ushort_t* __restrict__ out) {
    size_t i = ((size_t)blockIdx.x * 256 + threadIdx.x) * 8;
    float4v a = *(const float4v*)(in + i);
    float4v b = *(const float4v*)(in + i + 4);
    ushort8v o;
    o[0]=f2bf(a.x); o[1]=f2bf(a.y); o[2]=f2bf(a.z); o[3]=f2bf(a.w);
    o[4]=f2bf(b.x); o[5]=f2bf(b.y); o[6]=f2bf(b.z); o[7]=f2bf(b.w);
    *(ushort8v*)(out + i) = o;
}

// ---------------------------------------------------------------------------
// 2) W_enc (K=ENC x N=ATTN) -> Wt bf16, N-major (each row of Wt = a column of
//    W_enc, K-contiguous) so WMMA B fragments are contiguous loads.
// ---------------------------------------------------------------------------
__global__ void wt_kernel(const float* __restrict__ W, ushort_t* __restrict__ Wt) {
    __shared__ ushort_t tile[32][33];
    int n0 = blockIdx.x * 32, k0 = blockIdx.y * 32;
    int tx = threadIdx.x & 31, ty = threadIdx.x >> 5;   // 256 threads: 32x8
    #pragma unroll
    for (int i = 0; i < 32; i += 8)
        tile[ty + i][tx] = f2bf(W[(size_t)(k0 + ty + i) * AA + n0 + tx]);
    __syncthreads();
    #pragma unroll
    for (int i = 0; i < 32; i += 8)
        Wt[(size_t)(n0 + ty + i) * EE + k0 + tx] = tile[tx][ty + i];
}

// ---------------------------------------------------------------------------
// 3) qv[b][a] = hidden @ W_hidden + b_attn   (tiny: 64 MFLOP, fp32 VALU)
// ---------------------------------------------------------------------------
__global__ void q_kernel(const float* __restrict__ hidden, const float* __restrict__ Wh,
                         const float* __restrict__ bA, float* __restrict__ qv) {
    int idx = blockIdx.x * 256 + threadIdx.x;
    int b = idx >> 10, a = idx & 1023;
    float acc = bA[a];
    for (int h = 0; h < 1024; ++h)
        acc = fmaf(hidden[b * 1024 + h], Wh[h * 1024 + a], acc);
    qv[idx] = acc;
}

// ---------------------------------------------------------------------------
// 4) Fused score GEMM: logits[m] = sum_n tanh(enc[m,:]@Wenc[:,n] + qv[b,n]) * v[n]
//    4 waves, each owning a 16-row tile. N processed in 32-column chunks:
//    the A fragment is loaded once per kt and reused for TWO WMMAs
//    (3 ds_load_b128 per wmma instead of 4 -> -25% LDS traffic, which is
//    the binding resource). W chunks double-buffered via TDM.
// ---------------------------------------------------------------------------
__global__ void __launch_bounds__(128, 1)
score_kernel(const ushort_t* __restrict__ encbf, const float* __restrict__ enc32,
             const ushort_t* __restrict__ wt, const float* __restrict__ qv,
             const float* __restrict__ vvec, float* __restrict__ logits, int preconv) {
    extern __shared__ char smem[];
    const uint_t ldsBase = (uint_t)(unsigned long long)(void*)smem;
    const int tid  = threadIdx.x;
    const int wave = tid >> 5, lane = tid & 31;
    const int nlo = lane & 15, hi = lane >> 4;
    const int mBase = blockIdx.x * MT;

    if (preconv) {
        if (wave == 0)   // one TENSOR_LOAD_TO_LDS: A tile 64x1024 bf16
            tdm_load_2d(ldsBase, encbf + (size_t)mBase * EE, EE, MT, EE, MT, EE);
    } else {
        // cooperative fp32->bf16 A-tile conversion into the padded LDS layout
        const float4v* src = (const float4v*)(enc32 + (size_t)mBase * EE);
        for (int it = 0; it < 128; ++it) {
            int j4 = it * 128 + tid;              // float4 index within tile
            float4v f = src[j4];
            int j = j4 * 4;
            uint_t row = (uint_t)(j >> 10);
            uint_t x   = (uint_t)(j & 1023) * 2u;
            ushort4v o;
            o[0]=f2bf(f.x); o[1]=f2bf(f.y); o[2]=f2bf(f.z); o[3]=f2bf(f.w);
            *(ushort4v*)(smem + row * ROWB + pad_off(x)) = o;
        }
    }
    if (wave == 0) {     // first W chunk: 32 cols x 1024 K bf16 from Wt
        tdm_load_2d(ldsBase + A_BYTES, wt, EE, 32, EE, 32, EE);
        __builtin_amdgcn_s_wait_tensorcnt(0);
    }
    __syncthreads();

    const uint_t rowOff  = (uint_t)(wave * 16 + nlo) * ROWB;  // A: lane -> row M
    const uint_t colBase = (uint_t)nlo * ROWB;                // B: lane -> col N
    v8f score = {0.f,0.f,0.f,0.f,0.f,0.f,0.f,0.f};

    for (int ntb = 0; ntb < 32; ++ntb) {          // 32-column N chunks
        if (wave == 0 && ntb + 1 < 32)            // prefetch next W chunk
            tdm_load_2d(ldsBase + A_BYTES + (uint_t)((ntb + 1) & 1) * BCH_BYTES,
                        wt + (size_t)(ntb + 1) * 32 * EE, EE, 32, EE, 32, EE);

        const uint_t bufOff = A_BYTES + (uint_t)(ntb & 1) * BCH_BYTES;
        v8f acc0 = {0.f,0.f,0.f,0.f,0.f,0.f,0.f,0.f};
        v8f acc1 = {0.f,0.f,0.f,0.f,0.f,0.f,0.f,0.f};
        #pragma unroll 4
        for (int kt = 0; kt < 32; ++kt) {
            // A-frag: lane<16 holds K {k0..k0+7, k0+16..k0+23}; lane>=16 shifted +8
            uint_t x = pad_off((uint_t)kt * 64u + (uint_t)hi * 16u);
            union { uint4v u[2]; v16bf h; } Af, Bf0, Bf1;
            Af.u[0]  = *(const uint4v*)(smem + rowOff + x);
            Af.u[1]  = *(const uint4v*)(smem + rowOff + x + 32u);
            Bf0.u[0] = *(const uint4v*)(smem + bufOff + colBase + x);
            Bf0.u[1] = *(const uint4v*)(smem + bufOff + colBase + x + 32u);
            Bf1.u[0] = *(const uint4v*)(smem + bufOff + colBase + 16u * ROWB + x);
            Bf1.u[1] = *(const uint4v*)(smem + bufOff + colBase + 16u * ROWB + x + 32u);
            acc0 = __builtin_amdgcn_wmma_f32_16x16x32_bf16(
                       false, Af.h, false, Bf0.h, (short)0, acc0, false, false);
            acc1 = __builtin_amdgcn_wmma_f32_16x16x32_bf16(
                       false, Af.h, false, Bf1.h, (short)0, acc1, false, false);
        }
        // epilogue: key + q -> tanh -> * v[n] for both 16-col subtiles
        #pragma unroll
        for (int s = 0; s < 2; ++s) {
            const int n = ntb * 32 + s * 16 + nlo;
            const float vn = vvec[n];
            const v8f& acc = s ? acc1 : acc0;
            #pragma unroll
            for (int r = 0; r < 8; ++r) {
                int m = mBase + wave * 16 + r + hi * 8;    // C layout: M = r + 8*hi
                float q = qv[((m & (BB - 1)) << 10) + n];  // b = m % 32
                score[r] += fast_tanh(acc[r] + q) * vn;
            }
        }
        if (wave == 0) __builtin_amdgcn_s_wait_tensorcnt(0);
        __syncthreads();
    }

    // reduce the 16 N-lanes of each half-wave; lane0/lane16 hold row sums
    #pragma unroll
    for (int r = 0; r < 8; ++r) {
        float s = score[r];
        for (int off = 1; off < 16; off <<= 1) s += __shfl_xor(s, off, 32);
        if (nlo == 0) logits[mBase + wave * 16 + r + hi * 8] = s;
    }
}

// ---------------------------------------------------------------------------
// 5) masked softmax over L per column b (logits laid out m = l*B + b)
// ---------------------------------------------------------------------------
__global__ void softmax_kernel(const float* __restrict__ logits, const float* __restrict__ mask,
                               float* __restrict__ alignOut) {
    __shared__ float red[256];
    const int b = blockIdx.x, t = threadIdx.x;
    float vals[8], mx = -1e30f;
    #pragma unroll
    for (int i = 0; i < 8; ++i) {
        int m = (t + i * 256) * BB + b;
        vals[i] = logits[m] + mask[m];
        mx = fmaxf(mx, vals[i]);
    }
    red[t] = mx; __syncthreads();
    for (int s = 128; s > 0; s >>= 1) { if (t < s) red[t] = fmaxf(red[t], red[t + s]); __syncthreads(); }
    mx = red[0]; __syncthreads();
    float sum = 0.f;
    #pragma unroll
    for (int i = 0; i < 8; ++i) { vals[i] = __expf(vals[i] - mx); sum += vals[i]; }
    red[t] = sum; __syncthreads();
    for (int s = 128; s > 0; s >>= 1) { if (t < s) red[t] += red[t + s]; __syncthreads(); }
    const float inv = 1.f / red[0];
    #pragma unroll
    for (int i = 0; i < 8; ++i) alignOut[(t + i * 256) * BB + b] = vals[i] * inv;
}

// ---------------------------------------------------------------------------
// 6) context[b][e] = sum_l align[l,b] * enc[l,b,e] — deterministic 2-stage
// ---------------------------------------------------------------------------
__global__ void context_part_kernel(const ushort_t* __restrict__ encbf, const float* __restrict__ enc32,
                                    const float* __restrict__ align, float* __restrict__ part,
                                    int preconv) {
    const int e = blockIdx.x * 256 + threadIdx.x;
    const int b = blockIdx.y;
    const int l0 = blockIdx.z * 128;
    float acc = 0.f;
    for (int i = 0; i < 128; ++i) {
        int m = (l0 + i) * BB + b;
        float x = preconv ? bf2f(encbf[(size_t)m * EE + e]) : enc32[(size_t)m * EE + e];
        acc = fmaf(align[m], x, acc);
    }
    part[(size_t)blockIdx.z * (BB * EE) + b * EE + e] = acc;
}

__global__ void context_reduce_kernel(const float* __restrict__ part, float* __restrict__ ctx) {
    const int e = blockIdx.x * 256 + threadIdx.x;
    const int b = blockIdx.y;
    float acc = 0.f;
    #pragma unroll
    for (int z = 0; z < 16; ++z) acc += part[(size_t)z * (BB * EE) + b * EE + e];
    ctx[b * EE + e] = acc;
}

// ---------------------------------------------------------------------------
extern "C" void kernel_launch(void* const* d_in, const int* in_sizes, int n_in,
                              void* d_out, int out_size, void* d_ws, size_t ws_size,
                              hipStream_t stream) {
    const float* enc   = (const float*)d_in[0];
    const float* mask  = (const float*)d_in[1];
    const float* hid   = (const float*)d_in[2];
    const float* Wenc  = (const float*)d_in[3];
    const float* battn = (const float*)d_in[4];
    const float* Whid  = (const float*)d_in[5];
    const float* v     = (const float*)d_in[6];
    float* ctx   = (float*)d_out;                 // (B, ENC)
    float* align = (float*)d_out + BB * EE;       // (L, B)

    const size_t ENCBF_BYTES = (size_t)MM * EE * 2;          // 128 MB
    const size_t WT_BYTES    = (size_t)AA * EE * 2;          // 2 MB
    const size_t QV_BYTES    = (size_t)BB * AA * 4;          // 128 KB
    const size_t LG_BYTES    = (size_t)MM * 4;               // 256 KB
    const size_t PART_BYTES  = (size_t)16 * BB * EE * 4;     // 2 MB
    const size_t NEED_BIG = ENCBF_BYTES + WT_BYTES + QV_BYTES + LG_BYTES + PART_BYTES;

    char* ws = (char*)d_ws;
    const int preconv = (ws_size >= NEED_BIG) ? 1 : 0;
    size_t off = 0;
    unsigned short* encbf = nullptr;
    if (preconv) { encbf = (unsigned short*)ws; off += ENCBF_BYTES; }
    unsigned short* wt = (unsigned short*)(ws + off); off += WT_BYTES;
    float* qv     = (float*)(ws + off); off += QV_BYTES;
    float* logits = (float*)(ws + off); off += LG_BYTES;
    float* part   = (float*)(ws + off);

    if (preconv)
        convert_bf16_kernel<<<(MM * EE) / (256 * 8), 256, 0, stream>>>(enc, encbf);
    wt_kernel<<<dim3(AA / 32, EE / 32), 256, 0, stream>>>(Wenc, wt);
    q_kernel<<<(BB * AA) / 256, 256, 0, stream>>>(hid, Whid, battn, qv);
    score_kernel<<<MM / MT, 128, SMEM_BYTES, stream>>>(encbf, enc, wt, qv, v, logits, preconv);
    softmax_kernel<<<BB, 256, 0, stream>>>(logits, mask, align);
    context_part_kernel<<<dim3(EE / 256, BB, 16), 256, 0, stream>>>(encbf, enc, align, part, preconv);
    context_reduce_kernel<<<dim3(EE / 256, BB), 256, 0, stream>>>(part, ctx);
}
